// Seq2SeqLuoAtt_31181462569446
// MI455X (gfx1250) — matrix operand order repaired and checked by reading, not compiled, not run on hardware
//
#include <hip/hip_runtime.h>
#include <math.h>

// Problem constants (from reference): B=64, T=256, C=2048, WE=256, H2=1024
constexpr int Bn  = 64;
constexpr int Tn  = 256;
constexpr int Cn  = 2048;
constexpr int WEn = 256;
constexpr int Hn  = 1024;
constexpr int Gn  = 3 * Hn;   // 3072

// ---------------------------------------------------------------------------
// WMMA helpers (CDNA5 gfx1250, wave32)
// ---------------------------------------------------------------------------
typedef __attribute__((ext_vector_type(16))) __bf16 v16bf;
typedef __attribute__((ext_vector_type(8)))  float  v8f;

union FragBF {
  v16bf v;
  uint4 q[2];
};

__device__ __forceinline__ v16bf load_frag(const __bf16* p0, const __bf16* p1) {
  FragBF f;
  f.q[0] = *reinterpret_cast<const uint4*>(p0);
  f.q[1] = *reinterpret_cast<const uint4*>(p1);
  return f.v;
}

// f32 -> bf16 round-to-nearest-even
__device__ __forceinline__ __bf16 f2bf(float x) {
  union { unsigned int u; float f; } in; in.f = x;
  unsigned int r = in.u + 0x7FFFu + ((in.u >> 16) & 1u);
  union { unsigned short s; __bf16 b; } out; out.s = (unsigned short)(r >> 16);
  return out.b;
}

// One 16(M)x64(N) strip per wave: 4 accumulators, A fragment reused across 4
// WMMAs per K-chunk (2 A loads + 8 B loads per 4 WMMAs = 2.5 loads/WMMA).
// Fragment loads are double-buffered one K-chunk ahead so the 4 WMMAs of
// chunk k overlap the loads of chunk k+1 (needs ~130 VGPRs; see
// __launch_bounds__(256,1) on callers).
// Computes X @ W.T with W stored row-major [N][K] (row stride == K here).
// arow: per-lane A row (lane&15); brow0: per-lane W row (nbase + lane&15);
// consecutive n-tiles are +16*K apart. hi = lane>>4.
template <int K>
__device__ __forceinline__ void gemm_strip4(const __bf16* arow, const __bf16* brow0,
                                            v8f acc[4], int hi) {
  const int ak = hi * 16;   // A: lanes 16..31 hold K = kb+16..kb+31
  const int bk = hi * 16;   // B: lanes 16..31 hold K = kb+16..kb+31

  // prologue: fragments for chunk 0
  v16bf a = load_frag(arow + ak, arow + ak + 8);
  v16bf b[4];
#pragma unroll
  for (int j = 0; j < 4; ++j) {
    const __bf16* br = brow0 + (size_t)j * 16 * K;
    b[j] = load_frag(br + bk, br + bk + 8);
  }

#pragma unroll 4
  for (int kb = 0; kb < K; kb += 32) {
    v16bf a_cur = a;
    v16bf b_cur0 = b[0], b_cur1 = b[1], b_cur2 = b[2], b_cur3 = b[3];
    const int kn = kb + 32;
    if (kn < K) {   // prefetch chunk k+1 before consuming chunk k
      a = load_frag(arow + kn + ak, arow + kn + ak + 8);
#pragma unroll
      for (int j = 0; j < 4; ++j) {
        const __bf16* br = brow0 + (size_t)j * 16 * K;
        b[j] = load_frag(br + kn + bk, br + kn + bk + 8);
      }
    }
    acc[0] = __builtin_amdgcn_wmma_f32_16x16x32_bf16(false, a_cur, false, b_cur0,
                                                     (short)0, acc[0], false, false);
    acc[1] = __builtin_amdgcn_wmma_f32_16x16x32_bf16(false, a_cur, false, b_cur1,
                                                     (short)0, acc[1], false, false);
    acc[2] = __builtin_amdgcn_wmma_f32_16x16x32_bf16(false, a_cur, false, b_cur2,
                                                     (short)0, acc[2], false, false);
    acc[3] = __builtin_amdgcn_wmma_f32_16x16x32_bf16(false, a_cur, false, b_cur3,
                                                     (short)0, acc[3], false, false);
  }
}

// ---------------------------------------------------------------------------
// Setup kernels (run once, fully parallel)
// ---------------------------------------------------------------------------
__global__ void k_convert_bf16(const float* __restrict__ in, __bf16* __restrict__ out, int n) {
  int i = blockIdx.x * blockDim.x + threadIdx.x;
  if (i < n) out[i] = f2bf(in[i]);
}

// h0 = enc[:, T-1, :]; also seed the bf16 h half of the [h, c] concat buffer.
__global__ void k_init_h(const float* __restrict__ enc, float* __restrict__ h,
                         __bf16* __restrict__ hcbf) {
  int i = blockIdx.x * blockDim.x + threadIdx.x;   // 0..Bn*Hn-1
  int b = i >> 10, j = i & (Hn - 1);
  float v = enc[(size_t)b * Tn * Hn + (size_t)(Tn - 1) * Hn + j];
  h[i] = v;
  hcbf[(size_t)b * (2 * Hn) + j] = f2bf(v);
}

// gi[t][b][g] = sum_e E[id(t,b)][e] * W_ih[g][e] + b_ih[g]
// Embedding gather fused into the A-fragment load. 16x64 strip per wave;
// grid = (Gn/64/8, (Tn*Bn)/16) = (6, 1024).
__global__ void __launch_bounds__(256, 1)
k_gi_gemm(const __bf16* __restrict__ Ebf, const __bf16* __restrict__ Wih,
          const float* __restrict__ b_ih, const int* __restrict__ targets,
          float* __restrict__ gi) {
  const int lane = threadIdx.x & 31;
  const int wave = threadIdx.x >> 5;
  const int hi   = lane >> 4;
  const int nbase = (blockIdx.x * 8 + wave) * 64;   // 0..3008
  const int rbase = blockIdx.y * 16;                // 0..16368

  // Gather: row r = t*Bn + b ; id = (t==0) ? 0 : targets[b][t-1]
  int r  = rbase + (lane & 15);
  int t  = r >> 6;
  int b  = r & (Bn - 1);
  int id = (t == 0) ? 0 : targets[b * Tn + (t - 1)];
  const __bf16* arow  = Ebf + (size_t)id * WEn;
  const __bf16* brow0 = Wih + (size_t)(nbase + (lane & 15)) * WEn;

  v8f acc[4] = {};
  gemm_strip4<WEn>(arow, brow0, acc, hi);

#pragma unroll
  for (int j = 0; j < 4; ++j) {
    int n = nbase + j * 16 + (lane & 15);
    float bias = b_ih[n];
#pragma unroll
    for (int i = 0; i < 8; ++i) {
      int row = rbase + i + hi * 8;
      gi[(size_t)row * Gn + n] = acc[j][i] + bias;
    }
  }
}

// ---------------------------------------------------------------------------
// Per-step kernels (serial chain, one stream)
// ---------------------------------------------------------------------------

// gh = h @ W_hh.T + b_hh   (M=64, N=3072, K=1024). A = bf16 h half of hcbf
// (row stride 2*Hn). grid = (6, 4).
__global__ void __launch_bounds__(256, 1)
k_gh_gemm(const __bf16* __restrict__ hcbf, const __bf16* __restrict__ Whh,
          const float* __restrict__ b_hh, float* __restrict__ gh) {
  const int lane = threadIdx.x & 31;
  const int wave = threadIdx.x >> 5;
  const int hi   = lane >> 4;
  const int nbase = (blockIdx.x * 8 + wave) * 64;   // 0..3008
  const int rbase = blockIdx.y * 16;                // 0..48

  const __bf16* arow  = hcbf + (size_t)(rbase + (lane & 15)) * (2 * Hn);
  const __bf16* brow0 = Whh + (size_t)(nbase + (lane & 15)) * Hn;

  v8f acc[4] = {};
  gemm_strip4<Hn>(arow, brow0, acc, hi);

#pragma unroll
  for (int j = 0; j < 4; ++j) {
    int n = nbase + j * 16 + (lane & 15);
    float bias = b_hh[n];
#pragma unroll
    for (int i = 0; i < 8; ++i) {
      int row = rbase + i + hi * 8;
      gh[(size_t)row * Gn + n] = acc[j][i] + bias;
    }
  }
}

// GRU gate math (f32), writes h (f32) and the bf16 h half of hcbf.
__global__ void k_gru(const float* __restrict__ gi_t, const float* __restrict__ gh,
                      float* __restrict__ h, __bf16* __restrict__ hcbf) {
  int i = blockIdx.x * blockDim.x + threadIdx.x;   // 0..Bn*Hn-1
  int b = i >> 10, j = i & (Hn - 1);
  const float* gib = gi_t + (size_t)b * Gn;
  const float* ghb = gh   + (size_t)b * Gn;
  float ir = gib[j], iz = gib[Hn + j], in = gib[2 * Hn + j];
  float hr = ghb[j], hz = ghb[Hn + j], hn = ghb[2 * Hn + j];
  float r = 1.0f / (1.0f + expf(-(ir + hr)));
  float z = 1.0f / (1.0f + expf(-(iz + hz)));
  float n = tanhf(in + r * hn);
  float hv = (1.0f - z) * n + z * h[i];
  h[i] = hv;
  hcbf[(size_t)b * (2 * Hn) + j] = f2bf(hv);
}

// Luong attention for one batch element per block (L2-resident encoder reads).
// Writes the bf16 context half of hcbf.
__global__ void __launch_bounds__(256)
k_attn(const float* __restrict__ enc, const unsigned char* __restrict__ mask,
       const float* __restrict__ h, __bf16* __restrict__ hcbf) {
  __shared__ float sh[Hn];
  __shared__ float se[Tn];
  __shared__ float sred[8];
  const int b    = blockIdx.x;
  const int tid  = threadIdx.x;
  const int lane = tid & 31;
  const int wave = tid >> 5;
  const float* encb = enc + (size_t)b * Tn * Hn;

  for (int j = tid; j < Hn; j += 256) sh[j] = h[(size_t)b * Hn + j];
  __syncthreads();

  // scores: e[t] = dot(h_new, enc[b,t,:]) ; one dot per wave iteration
  for (int t = wave; t < Tn; t += 8) {
    const float* row = encb + (size_t)t * Hn;
    float s = 0.0f;
    for (int j = lane; j < Hn; j += 32) s += sh[j] * row[j];
    for (int off = 16; off > 0; off >>= 1) s += __shfl_down(s, off, 32);
    if (lane == 0) se[t] = mask[b * Tn + t] ? s : -INFINITY;
  }
  __syncthreads();

  // masked softmax over T=256 (one value per thread)
  float eval = se[tid];
  float v = eval;
  for (int off = 16; off > 0; off >>= 1) v = fmaxf(v, __shfl_down(v, off, 32));
  if (lane == 0) sred[wave] = v;
  __syncthreads();
  float gmax = -INFINITY;
#pragma unroll
  for (int w = 0; w < 8; ++w) gmax = fmaxf(gmax, sred[w]);
  float ex = expf(eval - gmax);
  float sv = ex;
  for (int off = 16; off > 0; off >>= 1) sv += __shfl_down(sv, off, 32);
  __syncthreads();
  if (lane == 0) sred[wave] = sv;
  __syncthreads();
  float gsum = 0.0f;
#pragma unroll
  for (int w = 0; w < 8; ++w) gsum += sred[w];
  se[tid] = ex / gsum;
  __syncthreads();

  // context: c[j] = sum_t alpha[t] * enc[b,t,j]; 4 dims per thread
  float cacc[4] = {0.f, 0.f, 0.f, 0.f};
  for (int t = 0; t < Tn; ++t) {
    float a = se[t];
    const float* row = encb + (size_t)t * Hn;
#pragma unroll
    for (int i = 0; i < 4; ++i) cacc[i] += a * row[tid + 256 * i];
  }
#pragma unroll
  for (int i = 0; i < 4; ++i)
    hcbf[(size_t)b * (2 * Hn) + Hn + tid + 256 * i] = f2bf(cacc[i]);
}

// logits[:, t, :] = [h, c] @ W_out.T + b_out  (M=64, N=2048, K=2048). grid (4,4).
__global__ void __launch_bounds__(256, 1)
k_logits(const __bf16* __restrict__ hcbf, const __bf16* __restrict__ Wout,
         const float* __restrict__ b_out, float* __restrict__ out, int t) {
  const int lane = threadIdx.x & 31;
  const int wave = threadIdx.x >> 5;
  const int hi   = lane >> 4;
  const int nbase = (blockIdx.x * 8 + wave) * 64;   // 0..1984
  const int rbase = blockIdx.y * 16;                // 0..48

  const __bf16* arow  = hcbf + (size_t)(rbase + (lane & 15)) * (2 * Hn);
  const __bf16* brow0 = Wout + (size_t)(nbase + (lane & 15)) * (2 * Hn);

  v8f acc[4] = {};
  gemm_strip4<2 * Hn>(arow, brow0, acc, hi);

#pragma unroll
  for (int j = 0; j < 4; ++j) {
    int n = nbase + j * 16 + (lane & 15);
    float bias = b_out[n];
#pragma unroll
    for (int i = 0; i < 8; ++i) {
      int row = rbase + i + hi * 8;   // batch index
      out[((size_t)row * Tn + t) * Cn + n] = acc[j][i] + bias;
    }
  }
}

// ---------------------------------------------------------------------------
// Host driver
// ---------------------------------------------------------------------------
extern "C" void kernel_launch(void* const* d_in, const int* in_sizes, int n_in,
                              void* d_out, int out_size, void* d_ws, size_t ws_size,
                              hipStream_t stream) {
  const float*         enc     = (const float*)d_in[0];
  const unsigned char* mask    = (const unsigned char*)d_in[1];   // jnp bool_ = 1 byte
  const int*           targets = (const int*)d_in[2];
  const float*         E       = (const float*)d_in[3];
  const float*         W_ih    = (const float*)d_in[4];
  const float*         W_hh    = (const float*)d_in[5];
  const float*         b_ih    = (const float*)d_in[6];
  const float*         b_hh    = (const float*)d_in[7];
  const float*         W_out   = (const float*)d_in[8];
  const float*         b_out   = (const float*)d_in[9];
  float*               out     = (float*)d_out;

  // workspace layout (256B aligned slabs), ~210 MB total
  char* ws = (char*)d_ws;
  size_t off = 0;
  auto alloc = [&](size_t bytes) -> void* {
    void* p = ws + off;
    off = (off + bytes + 255) & ~(size_t)255;
    return p;
  };
  __bf16* Ebf   = (__bf16*)alloc((size_t)Cn * WEn * 2);
  __bf16* Wihb  = (__bf16*)alloc((size_t)Gn * WEn * 2);
  __bf16* Whhb  = (__bf16*)alloc((size_t)Gn * Hn * 2);
  __bf16* Woutb = (__bf16*)alloc((size_t)Cn * (2 * Hn) * 2);
  float*  gi    = (float*)alloc((size_t)Tn * Bn * Gn * 4);
  float*  gh    = (float*)alloc((size_t)Bn * Gn * 4);
  float*  h     = (float*)alloc((size_t)Bn * Hn * 4);
  __bf16* hcbf  = (__bf16*)alloc((size_t)Bn * (2 * Hn) * 2);
  (void)ws_size; (void)in_sizes; (void)n_in; (void)out_size;

  // one-time: weight conversion, h0 init, gi precompute GEMM
  auto cvt = [&](const float* src, __bf16* dst, int n) {
    k_convert_bf16<<<(n + 255) / 256, 256, 0, stream>>>(src, dst, n);
  };
  cvt(E,     Ebf,   Cn * WEn);
  cvt(W_ih,  Wihb,  Gn * WEn);
  cvt(W_hh,  Whhb,  Gn * Hn);
  cvt(W_out, Woutb, Cn * 2 * Hn);
  k_init_h<<<(Bn * Hn) / 256, 256, 0, stream>>>(enc, h, hcbf);
  k_gi_gemm<<<dim3(Gn / 64 / 8, (Tn * Bn) / 16), 256, 0, stream>>>(
      Ebf, Wihb, b_ih, targets, gi);

  // serial 256-step chain: 4 kernels per step on one stream
  for (int t = 0; t < Tn; ++t) {
    k_gh_gemm<<<dim3(Gn / 64 / 8, Bn / 16), 256, 0, stream>>>(hcbf, Whhb, b_hh, gh);
    k_gru<<<(Bn * Hn) / 256, 256, 0, stream>>>(gi + (size_t)t * Bn * Gn, gh, h, hcbf);
    k_attn<<<Bn, 256, 0, stream>>>(enc, mask, h, hcbf);
    k_logits<<<dim3(Cn / 64 / 8, Bn / 16), 256, 0, stream>>>(hcbf, Woutb, b_out, out, t);
  }
}
